// Algin_81303730913442
// MI455X (gfx1250) — matrix operand order repaired and checked by reading, not compiled, not run on hardware
//
#include <hip/hip_runtime.h>
#include <cstdint>

typedef __attribute__((ext_vector_type(16))) _Float16 v16h;
typedef __attribute__((ext_vector_type(8)))  float    v8f;
typedef long long i64;

#define CDIV(a,b) (((a)+(b)-1)/(b))

// ---------------------------------------------------------------------------
// Weight packing into WMMA B-fragment layout.
// B fragment (16x16x32 f16): lane L -> N = L&15, K(within chunk) = (e&7) + (L>>4)*8 + (e>>3)*16
// Stored as pk[(((chunk*NTILE + ntile)*32 + lane)*16 + e)]
// ---------------------------------------------------------------------------

// conv2/conv3 weights: OIHW (16,16,3,3). K-order = tap*16 + c, 5 chunks (K=144->160).
__global__ __launch_bounds__(256) void pack_w16_kernel(const float* __restrict__ w,
                                                       _Float16* __restrict__ pk) {
  int i = blockIdx.x * 256 + threadIdx.x;
  if (i >= 5 * 32 * 16) return;
  int e = i & 15, lane = (i >> 4) & 31, cc = i >> 9;
  int k = cc * 32 + ((e >> 3) << 4) + ((lane >> 4) << 3) + (e & 7);
  int n = lane & 15;
  float v = 0.f;
  if (k < 144) { int tap = k >> 4, c = k & 15; v = w[(n * 16 + c) * 9 + tap]; }
  pk[i] = (_Float16)v;
}

// offset-conv weights: OIHW (144,32,3,3). K-order = tap*32 + c, 9 chunks, 9 N-tiles.
__global__ __launch_bounds__(256) void pack_w32_kernel(const float* __restrict__ w,
                                                       _Float16* __restrict__ pk) {
  int i = blockIdx.x * 256 + threadIdx.x;
  if (i >= 9 * 9 * 32 * 16) return;
  int e = i & 15, lane = (i >> 4) & 31;
  int g = i >> 9; int nt = g % 9, cc = g / 9;
  int c = ((e >> 3) << 4) + ((lane >> 4) << 3) + (e & 7);
  int n = nt * 16 + (lane & 15);
  pk[i] = (_Float16)w[(n * 32 + c) * 9 + cc];
}

// ---------------------------------------------------------------------------
// conv1: 3->16, 3x3, pad 1, relu. NCHW f32 in, NHWC f16 out.
// grid = (CDIV(W,256), B*H); blockIdx.y = b*H + y (row decomposition is scalar).
// ---------------------------------------------------------------------------
__global__ __launch_bounds__(256) void conv1_kernel(const float* __restrict__ img,
    const float* __restrict__ w, const float* __restrict__ bias,
    _Float16* __restrict__ outF, int H, int W) {
  __shared__ float ws[16 * 27];
  for (int i = threadIdx.x; i < 16 * 27; i += 256) ws[i] = w[i];
  __syncthreads();
  const int x = blockIdx.x * 256 + threadIdx.x;
  if (x >= W) return;
  const int row = blockIdx.y;            // b*H + y
  const int y = row % H, b = row / H;    // SALU
  float in[27];
#pragma unroll
  for (int ci = 0; ci < 3; ++ci)
#pragma unroll
    for (int tt = 0; tt < 9; ++tt) {
      int iy = y + tt / 3 - 1, ix = x + tt % 3 - 1;
      in[ci * 9 + tt] = (iy >= 0 && iy < H && ix >= 0 && ix < W)
                          ? img[((b * 3 + ci) * H + iy) * W + ix] : 0.f;
    }
  alignas(16) _Float16 ov[16];
#pragma unroll
  for (int o = 0; o < 16; ++o) {
    float a = bias[o];
#pragma unroll
    for (int j = 0; j < 27; ++j) a += ws[o * 27 + j] * in[j];
    ov[o] = (_Float16)fmaxf(a, 0.f);
  }
  uint4* dst = (uint4*)(outF + (row * W + x) * 16);
  dst[0] = ((uint4*)ov)[0];
  dst[1] = ((uint4*)ov)[1];
}

// ---------------------------------------------------------------------------
// Generic 3x3 conv as WMMA GEMM. Each wave: 16 pixels (one row segment) x NTILE*16 ch.
// CIN==32: two 16ch f16 NHWC buffers (channel concat), 1 tap per 32-K chunk.
// CIN==16: single input, 2 taps per chunk, 5 chunks (K=144 zero-padded to 160).
// A-fragment (ISA 7.12.2, 16-bit A 16x32): lane L row = L&15;
//   elements 0..7 -> K = e + (L>>4)*8, elements 8..15 -> K = (e-8) + 16 + (L>>4)*8.
// Channel-fastest K ordering => each 8-element half is one contiguous b128 load.
// All indexing 32-bit; tile/row decomposition forced scalar via readfirstlane.
// ---------------------------------------------------------------------------
template <int CIN, int NTILE, bool OUT32>
__global__ __launch_bounds__(256) void conv_wmma_kernel(
    const _Float16* __restrict__ inA, const _Float16* __restrict__ inB2,
    const _Float16* __restrict__ pack, const float* __restrict__ bias,
    void* __restrict__ outp, int Batch, int Hin, int Win,
    int Hout, int Wout, int pad, int relu) {
  constexpr int NCHUNK = (CIN == 16) ? 5 : 9;
  constexpr int COUT = NTILE * 16;
  const int lane = threadIdx.x & 31;
  // wave id is uniform within the wave -> pin it to an SGPR so tile math is SALU
  const int wave = __builtin_amdgcn_readfirstlane(threadIdx.x >> 5);
  const int tilesPerRow = (Wout + 15) >> 4;
  const int nRows = Batch * Hout;
  const int tileId = blockIdx.x * 8 + wave;           // scalar
  const int row = tileId / tilesPerRow;               // scalar div
  if (row >= nRows) return;                           // wave-uniform: EXEC all-1 at WMMA
  const int xBase = (tileId - row * tilesPerRow) << 4;
  const int y = row % Hout;                           // scalar
  const int b = row / Hout;                           // scalar
  __builtin_prefetch(pack, 0, 3);
  const int kh = lane >> 4;
  const int c0 = kh * 8;
  const int m = lane & 15;
  int x = xBase + m;
  if (x >= Wout) x = Wout - 1;                        // clamp; these rows never stored

  union Frag { v16h v; uint4 q[2]; };
  Frag A[NCHUNK];
  const uint4 z4 = make_uint4(0u, 0u, 0u, 0u);
#pragma unroll
  for (int cc = 0; cc < NCHUNK; ++cc) {
    if constexpr (CIN == 32) {
      const int iy = y + cc / 3 - pad;                // scalar
      const int ix = x + cc % 3 - pad;                // per-lane
      if (iy >= 0 && iy < Hin && ix >= 0 && ix < Win) {
        const int base = (((b * Hin + iy) * Win + ix) << 4) + c0;
        A[cc].q[0] = *(const uint4*)(inA + base);     // cat channels 0..15  (ref)
        A[cc].q[1] = *(const uint4*)(inB2 + base);    // cat channels 16..31 (unr)
      } else {
        A[cc].q[0] = z4; A[cc].q[1] = z4;
      }
    } else {
      const int t0 = 2 * cc, t1 = 2 * cc + 1;
      {
        const int iy = y + t0 / 3 - pad, ix = x + t0 % 3 - pad;
        A[cc].q[0] = (iy >= 0 && iy < Hin && ix >= 0 && ix < Win)
          ? *(const uint4*)(inA + (((b * Hin + iy) * Win + ix) << 4) + c0) : z4;
      }
      if (t1 < 9) {
        const int iy = y + t1 / 3 - pad, ix = x + t1 % 3 - pad;
        A[cc].q[1] = (iy >= 0 && iy < Hin && ix >= 0 && ix < Win)
          ? *(const uint4*)(inA + (((b * Hin + iy) * Win + ix) << 4) + c0) : z4;
      } else {
        A[cc].q[1] = z4;                              // K tail padding (tap 9)
      }
    }
  }

  const int pixBase = row * Wout;
#pragma unroll
  for (int nt = 0; nt < NTILE; ++nt) {
    v8f acc = {};
#pragma unroll
    for (int cc = 0; cc < NCHUNK; ++cc) {
      Frag Bf;
      const _Float16* bp = pack + (((cc * NTILE + nt) * 32 + lane) << 4);
      Bf.q[0] = *(const uint4*)(bp);
      Bf.q[1] = *(const uint4*)(bp + 8);
      acc = __builtin_amdgcn_wmma_f32_16x16x32_f16(false, A[cc].v, false, Bf.v,
                                                   (short)0, acc, false, false);
    }
    const int n = nt * 16 + m;
    const float bv = bias[n];
#pragma unroll
    for (int r = 0; r < 8; ++r) {       // D layout: lane L, VGPR r -> M=r+(L>>4)*8, N=L&15
      const int xx = xBase + r + kh * 8;
      if (xx < Wout) {
        float v = acc[r] + bv;
        if (relu) v = fmaxf(v, 0.f);
        const int idx = (pixBase + xx) * COUT + n;
        if constexpr (OUT32)
          ((float*)outp)[idx] = v;
        else
          ((_Float16*)outp)[idx] = (_Float16)v;
      }
    }
  }
}

// ---------------------------------------------------------------------------
// Deformable conv: feat = unr_f (f16 NHWC, 16ch), off = f32 [pixel][144]
// (channel order (g*9+k)*2 + {y,x}), dcn_w (16,16,3,3) f32, out NCHW f32.
// grid = (CDIV(W,256), B*H); blockIdx.y = b*H + y.
// ---------------------------------------------------------------------------
__global__ __launch_bounds__(256) void deform_kernel(
    const _Float16* __restrict__ feat, const float* __restrict__ off,
    const float* __restrict__ w, const float* __restrict__ bias,
    float* __restrict__ out, int H, int W) {
  __shared__ float ws[16 * 144];
  for (int i = threadIdx.x; i < 16 * 144; i += 256) ws[i] = w[i];
  __syncthreads();
  const int x = blockIdx.x * 256 + threadIdx.x;
  if (x >= W) return;
  const int row = blockIdx.y;            // b*H + y
  const int y = row % H, bb = row / H;   // SALU
  float acc[16];
#pragma unroll
  for (int o = 0; o < 16; ++o) acc[o] = bias[o];
  const float* offp = off + (row * W + x) * 144;
  const _Float16* fb = feat + bb * H * W * 16;
  for (int g = 0; g < 8; ++g) {
    const int cbase = g * 2;
#pragma unroll
    for (int k = 0; k < 9; ++k) {
      float oy = offp[(g * 9 + k) * 2 + 0];
      float ox = offp[(g * 9 + k) * 2 + 1];
      float ysf = (float)y - 1.0f + (float)(k / 3) + oy;
      float xsf = (float)x - 1.0f + (float)(k % 3) + ox;
      float y0 = floorf(ysf), x0 = floorf(xsf);
      float dy = ysf - y0, dx = xsf - x0;
      int y0i = (int)y0, x0i = (int)x0;
      float v0 = 0.f, v1 = 0.f;
#pragma unroll
      for (int cy = 0; cy < 2; ++cy)
#pragma unroll
        for (int cx = 0; cx < 2; ++cx) {
          int iy = y0i + cy, ix = x0i + cx;
          if (iy >= 0 && iy < H && ix >= 0 && ix < W) {
            float wgt = (cy ? dy : 1.f - dy) * (cx ? dx : 1.f - dx);
            const _Float16* fp = fb + ((iy * W + ix) * 16 + cbase);
            v0 += wgt * (float)fp[0];
            v1 += wgt * (float)fp[1];
          }
        }
      const float* wrow = ws + (cbase * 9 + k);
#pragma unroll
      for (int o = 0; o < 16; ++o)
        acc[o] += wrow[o * 144] * v0 + wrow[o * 144 + 9] * v1;
    }
  }
  const int ob = bb * 16 * H * W + y * W + x;
#pragma unroll
  for (int o = 0; o < 16; ++o) out[ob + o * H * W] = acc[o];
}

// ---------------------------------------------------------------------------
extern "C" void kernel_launch(void* const* d_in, const int* in_sizes, int n_in,
                              void* d_out, int out_size, void* d_ws, size_t ws_size,
                              hipStream_t stream) {
  const float* ref_img = (const float*)d_in[0];
  const float* unr_img = (const float*)d_in[1];
  const float* conv1_w = (const float*)d_in[2];
  const float* conv1_b = (const float*)d_in[3];
  const float* conv2_w = (const float*)d_in[4];
  const float* conv2_b = (const float*)d_in[5];
  const float* conv3_w = (const float*)d_in[6];
  const float* conv3_b = (const float*)d_in[7];
  const float* offw[3] = {(const float*)d_in[8], (const float*)d_in[10], (const float*)d_in[12]};
  const float* offb[3] = {(const float*)d_in[9], (const float*)d_in[11], (const float*)d_in[13]};
  const float* dcn_w = (const float*)d_in[14];
  const float* dcn_b = (const float*)d_in[15];
  float* out = (float*)d_out;

  const int B = 4, H1 = 256, H2 = 254, H3 = 252;
  const i64 n1 = (i64)B * H1 * H1, n2 = (i64)B * H2 * H2, n3 = (i64)B * H3 * H3;

  char* cur = (char*)d_ws;
  auto take = [&](size_t bytes) {
    char* r = cur;
    cur += (bytes + 255) & ~(size_t)255;
    return r;
  };
  _Float16* refF1 = (_Float16*)take(n1 * 16 * 2);
  _Float16* unrF1 = (_Float16*)take(n1 * 16 * 2);
  _Float16* refF2 = (_Float16*)take(n2 * 16 * 2);
  _Float16* unrF2 = (_Float16*)take(n2 * 16 * 2);
  _Float16* refF3 = (_Float16*)take(n3 * 16 * 2);
  _Float16* unrF3 = (_Float16*)take(n3 * 16 * 2);
  float*    offBuf = (float*)take(n1 * 144 * 4);   // reused across the 3 levels (L2-resident)
  _Float16* pkC2 = (_Float16*)take(5 * 32 * 16 * 2);
  _Float16* pkC3 = (_Float16*)take(5 * 32 * 16 * 2);
  _Float16* pkOff[3];
  for (int i = 0; i < 3; ++i) pkOff[i] = (_Float16*)take(9 * 9 * 32 * 16 * 2);

  pack_w16_kernel<<<10, 256, 0, stream>>>(conv2_w, pkC2);
  pack_w16_kernel<<<10, 256, 0, stream>>>(conv3_w, pkC3);
  for (int i = 0; i < 3; ++i)
    pack_w32_kernel<<<162, 256, 0, stream>>>(offw[i], pkOff[i]);

  conv1_kernel<<<dim3(CDIV(H1, 256), B * H1), 256, 0, stream>>>(ref_img, conv1_w, conv1_b, refF1, H1, H1);
  conv1_kernel<<<dim3(CDIV(H1, 256), B * H1), 256, 0, stream>>>(unr_img, conv1_w, conv1_b, unrF1, H1, H1);

  auto wmmaBlocks = [](int Batch, int Hout, int Wout) {
    return (Batch * Hout * ((Wout + 15) / 16) + 7) / 8;
  };
  conv_wmma_kernel<16, 1, false><<<wmmaBlocks(B, H2, H2), 256, 0, stream>>>(
      refF1, nullptr, pkC2, conv2_b, refF2, B, H1, H1, H2, H2, 0, 1);
  conv_wmma_kernel<16, 1, false><<<wmmaBlocks(B, H2, H2), 256, 0, stream>>>(
      unrF1, nullptr, pkC2, conv2_b, unrF2, B, H1, H1, H2, H2, 0, 1);
  conv_wmma_kernel<16, 1, false><<<wmmaBlocks(B, H3, H3), 256, 0, stream>>>(
      refF2, nullptr, pkC3, conv3_b, refF3, B, H2, H2, H3, H3, 0, 1);
  conv_wmma_kernel<16, 1, false><<<wmmaBlocks(B, H3, H3), 256, 0, stream>>>(
      unrF2, nullptr, pkC3, conv3_b, unrF3, B, H2, H2, H3, H3, 0, 1);

  const _Float16* refF[3] = {refF1, refF2, refF3};
  const _Float16* unrF[3] = {unrF1, unrF2, unrF3};
  const int Hs[3] = {H1, H2, H3};
  i64 outOff = 0;
  for (int i = 0; i < 3; ++i) {
    const int Hi = Hs[i];
    conv_wmma_kernel<32, 9, true><<<wmmaBlocks(B, Hi, Hi), 256, 0, stream>>>(
        refF[i], unrF[i], pkOff[i], offb[i], offBuf, B, Hi, Hi, Hi, Hi, 1, 1);
    deform_kernel<<<dim3(CDIV(Hi, 256), B * Hi), 256, 0, stream>>>(
        unrF[i], offBuf, dcn_w, dcn_b, out + outOff, Hi, Hi);
    outOff += (i64)B * 16 * Hi * Hi;
  }
  (void)in_sizes; (void)n_in; (void)out_size; (void)ws_size;
}
